// WindowAttention_12704513262110
// MI455X (gfx1250) — matrix operand order repaired and checked by reading, not compiled, not run on hardware
//
#include <hip/hip_runtime.h>
#include <hip/hip_bf16.h>

typedef __attribute__((ext_vector_type(16))) __bf16       bf16x16;
typedef __attribute__((ext_vector_type(8)))  float        f32x8;
typedef __attribute__((ext_vector_type(8)))  unsigned int u32x8;
typedef __attribute__((ext_vector_type(4)))  unsigned int u32x4;

#define WIN   49          // tokens per window
#define MPAD  64          // padded rows
#define C     128         // channels
#define H     4           // heads
#define HD    32          // head dim
#define Q3    384         // 3*C
#define SCALE 0.1767766952966369f   // 32^-0.5

// LDS strides (bf16 elements)
#define XB_S   130        // 65 dwords (odd -> conflict-free A rows)
#define WL_S   130
#define QK_S   260
#define VT_S   72         // 36 dwords; 16-byte aligned rows for b128 stores
#define P_S    66         // 33 dwords (odd)
#define O_S    130

#define XB_N   (MPAD * XB_S)
#define WL_N   (Q3   * WL_S)
#define QK_N   (MPAD * QK_S)
#define VT_N   (C    * VT_S)
#define P_N    (H * MPAD * P_S)
#define O_N    (MPAD * O_S)
#define SMEM_BYTES ((XB_N + WL_N + QK_N + VT_N + P_N + O_N) * 2 + (Q3 + C) * 4)

__device__ __forceinline__ unsigned short f2bf(float f) {
  unsigned int u = __builtin_bit_cast(unsigned int, f);
  u += 0x7FFFu + ((u >> 16) & 1u);            // round-to-nearest-even
  return (unsigned short)(u >> 16);
}
__device__ __forceinline__ unsigned int pk2bf(float lo, float hi) {
  return (unsigned int)f2bf(lo) | ((unsigned int)f2bf(hi) << 16);
}

// A operand: 16x32 bf16 tile, rows m0.., K k0..k0+31, from row-major [row][k]
__device__ __forceinline__ bf16x16 load_A(const unsigned short* base, int stride,
                                          int m0, int k0, int lane) {
  const unsigned short* p = base + (m0 + (lane & 15)) * stride + k0 + (lane >> 4) * 8;
  u32x8 u;
#pragma unroll
  for (int g = 0; g < 8; ++g) {
    const int koff = 2 * (g & 3) + ((g & 4) ? 16 : 0);
    u[g] = *(const unsigned int*)(p + koff);
  }
  return __builtin_bit_cast(bf16x16, u);
}

// B operand: 32x16 bf16 tile; element (k, n) = base[(n0+n)*stride + k0+k]
__device__ __forceinline__ bf16x16 load_B(const unsigned short* base, int stride,
                                          int n0, int k0, int lane) {
  const unsigned short* p = base + (n0 + (lane & 15)) * stride + k0 + (lane >> 4) * 8;
  u32x8 u;
#pragma unroll
  for (int g = 0; g < 8; ++g) u[g] = *(const unsigned int*)(p + 2 * (g & 3) + ((g & 4) ? 16 : 0));
  return __builtin_bit_cast(bf16x16, u);
}

__device__ __forceinline__ f32x8 wmma_bf16(bf16x16 a, bf16x16 b, f32x8 c) {
  return __builtin_amdgcn_wmma_f32_16x16x32_bf16(false, a, false, b, (short)0, c,
                                                 false, false);
}

__global__ __launch_bounds__(256, 1)
void win_attn_kernel(const float* __restrict__ x, const float* __restrict__ mask,
                     const float* __restrict__ qkv_w, const float* __restrict__ qkv_b,
                     const float* __restrict__ proj_w, const float* __restrict__ proj_b,
                     float* __restrict__ out) {
  extern __shared__ __align__(16) char smem[];
  unsigned short* Xb = (unsigned short*)smem;
  unsigned short* WL = Xb + XB_N;
  unsigned short* QK = WL + WL_N;
  unsigned short* Vt = QK + QK_N;
  unsigned short* Pm = Vt + VT_N;
  unsigned short* Ob = Pm + P_N;
  float*          qb = (float*)(Ob + O_N);
  float*          pb = qb + Q3;

  const int tid  = threadIdx.x;
  const int lane = tid & 31;
  const int wave = tid >> 5;
  const int blk  = blockIdx.x;
  const int colb = lane & 15;
  const int rb   = (lane >> 4) * 8;

  // ---- Phase 0: stage biases, x tile (bf16, zero-padded), qkv_w (bf16) ----
  if (tid < Q3) qb[tid] = qkv_b[tid];
  if (tid < C)  pb[tid] = proj_b[tid];
  const float2* xw2 = (const float2*)(x + (size_t)blk * WIN * C);
  for (int i = tid; i < MPAD * (C / 2); i += 256) {
    const int r = i >> 6, c2 = i & 63;
    float2 v = {0.0f, 0.0f};
    if (r < WIN) v = xw2[r * 64 + c2];
    *(unsigned int*)(Xb + r * XB_S + 2 * c2) = pk2bf(v.x, v.y);
  }
  const float2* w2 = (const float2*)qkv_w;
  for (int i = tid; i < Q3 * (C / 2); i += 256) {
    const float2 v = w2[i];
    *(unsigned int*)(WL + (i >> 6) * WL_S + 2 * (i & 63)) = pk2bf(v.x, v.y);
  }
  __syncthreads();

  // ---- Phase 1: QKV = X @ W^T + b   (96 tiles of 16x16, k = 4 x 32) ----
  for (int t = wave; t < 96; t += 8) {
    const int n0 = (t >> 2) * 16, m0 = (t & 3) * 16;
    f32x8 acc = {};
#pragma unroll
    for (int k0 = 0; k0 < C; k0 += 32)
      acc = wmma_bf16(load_A(Xb, XB_S, m0, k0, lane),
                      load_B(WL, WL_S, n0, k0, lane), acc);
    const int   col  = n0 + colb;
    const float bias = qb[col];
    if (n0 < C) {                                   // Q (pre-scaled)
#pragma unroll
      for (int i = 0; i < 8; ++i)
        QK[(m0 + rb + i) * QK_S + col] = f2bf((acc[i] + bias) * SCALE);
    } else if (n0 < 2 * C) {                        // K
#pragma unroll
      for (int i = 0; i < 8; ++i)
        QK[(m0 + rb + i) * QK_S + col] = f2bf(acc[i] + bias);
    } else {                                        // V, stored transposed, b128
      u32x4 w;
#pragma unroll
      for (int j = 0; j < 4; ++j)
        w[j] = pk2bf(acc[2 * j] + bias, acc[2 * j + 1] + bias);
      *(u32x4*)(Vt + (col - 2 * C) * VT_S + m0 + rb) = w;
    }
  }
  __syncthreads();

  // restage proj_w into the now-dead qkv_w region (read in phase 4)
  const float2* pw2 = (const float2*)proj_w;
  for (int i = tid; i < C * (C / 2); i += 256) {
    const float2 v = pw2[i];
    *(unsigned int*)(WL + (i >> 6) * WL_S + 2 * (i & 63)) = pk2bf(v.x, v.y);
  }

  // ---- Phase 2: S = Q @ K^T + mask, row softmax, P bf16 ----
  const float* mrow = mask + (size_t)(blk & 63) * WIN * WIN;
  for (int s = wave; s < 16; s += 8) {
    const int h = s >> 2, m0 = (s & 3) * 16;
    f32x8 acc[4];
    const bf16x16 a = load_A(QK, QK_S, m0, h * HD, lane);
#pragma unroll
    for (int j = 0; j < 4; ++j) {
      f32x8 z = {};
      acc[j] = wmma_bf16(a, load_B(QK + C + h * HD, QK_S, j * 16, 0, lane), z);
    }
#pragma unroll
    for (int j = 0; j < 4; ++j) {
      const int col = j * 16 + colb;
#pragma unroll
      for (int i = 0; i < 8; ++i) {
        const int row = m0 + rb + i;
        float v = acc[j][i];
        if (col >= WIN)      v = -1.0e30f;
        else if (row < WIN)  v += mrow[row * WIN + col];
        acc[j][i] = v;
      }
    }
#pragma unroll
    for (int i = 0; i < 8; ++i) {
      float m = fmaxf(fmaxf(acc[0][i], acc[1][i]), fmaxf(acc[2][i], acc[3][i]));
#pragma unroll
      for (int off = 8; off; off >>= 1) m = fmaxf(m, __shfl_xor(m, off, 16));
      const float e0 = __expf(acc[0][i] - m), e1 = __expf(acc[1][i] - m);
      const float e2 = __expf(acc[2][i] - m), e3 = __expf(acc[3][i] - m);
      float sum = e0 + e1 + e2 + e3;
#pragma unroll
      for (int off = 8; off; off >>= 1) sum += __shfl_xor(sum, off, 16);
      const float inv = 1.0f / sum;
      unsigned short* pr = Pm + (h * MPAD + m0 + rb + i) * P_S + colb;
      pr[0]  = f2bf(e0 * inv); pr[16] = f2bf(e1 * inv);
      pr[32] = f2bf(e2 * inv); pr[48] = f2bf(e3 * inv);
    }
  }
  __syncthreads();

  // ---- Phase 3: O = P @ V   (per head, k = 2 x 32) ----
  for (int t = wave; t < 32; t += 8) {
    const int h = t >> 3, mt = (t & 7) >> 1, dt = t & 1;
    const int m0 = mt * 16, d0 = h * HD + dt * 16;
    f32x8 acc = {};
#pragma unroll
    for (int k0 = 0; k0 < MPAD; k0 += 32)
      acc = wmma_bf16(load_A(Pm + h * MPAD * P_S, P_S, m0, k0, lane),
                      load_B(Vt, VT_S, d0, k0, lane), acc);
#pragma unroll
    for (int i = 0; i < 8; ++i)
      Ob[(m0 + rb + i) * O_S + d0 + colb] = f2bf(acc[i]);
  }
  __syncthreads();

  // ---- Phase 4: out = O @ proj_w^T + b, store fp32 rows < 49 ----
  float* ow = out + (size_t)blk * WIN * C;
  for (int t = wave; t < 32; t += 8) {
    const int m0 = (t >> 3) * 16, n0 = (t & 7) * 16;
    f32x8 acc = {};
#pragma unroll
    for (int k0 = 0; k0 < C; k0 += 32)
      acc = wmma_bf16(load_A(Ob, O_S, m0, k0, lane),
                      load_B(WL, WL_S, n0, k0, lane), acc);
    const int   col  = n0 + colb;
    const float bias = pb[col];
#pragma unroll
    for (int i = 0; i < 8; ++i) {
      const int row = m0 + rb + i;
      if (row < WIN) ow[row * C + col] = acc[i] + bias;
    }
  }
}

extern "C" void kernel_launch(void* const* d_in, const int* in_sizes, int n_in,
                              void* d_out, int out_size, void* d_ws, size_t ws_size,
                              hipStream_t stream) {
  const float* x      = (const float*)d_in[0];
  const float* mask   = (const float*)d_in[1];
  const float* qkv_w  = (const float*)d_in[2];
  const float* qkv_b  = (const float*)d_in[3];
  const float* proj_w = (const float*)d_in[4];
  const float* proj_b = (const float*)d_in[5];
  win_attn_kernel<<<4096, 256, SMEM_BYTES, stream>>>(x, mask, qkv_w, qkv_b,
                                                     proj_w, proj_b, (float*)d_out);
}
